// CrossAttention_4054449127644
// MI455X (gfx1250) — matrix operand order repaired and checked by reading, not compiled, not run on hardware
//
#include <hip/hip_runtime.h>
#include <hip/hip_bf16.h>

// CrossAttention for MI455X (gfx1250, wave32, WMMA + TDM).
// Pipeline: rmsnorm(x)->f16, rmsnorm(ctx)->f16, weight f32->f16 transpose,
// WMMA GEMM (Q), WMMA GEMM (KV), flash-style WMMA attention with TDM-staged
// Q/K tiles, WMMA GEMM (+bias), fused rmsnorm+residual epilogue.

#define B_     8
#define C_     512
#define HW_    1024            // 32*32
#define LC_    1024
#define HEADS_ 8
#define HD_    64
#define HID_   512             // HEADS_*HD_
#define NPIX_  (B_ * HW_)      // 8192
#define NTOK_  (B_ * LC_)      // 8192
#define EPS_   1e-6f

#define USE_TDM 1              // Tensor Data Mover staging of Q/K tiles

typedef __attribute__((ext_vector_type(16))) _Float16 v16h;
typedef __attribute__((ext_vector_type(8)))  _Float16 v8h;
typedef __attribute__((ext_vector_type(8)))  float    v8f;
typedef __attribute__((ext_vector_type(4)))  unsigned int su4;
typedef __attribute__((ext_vector_type(8)))  unsigned int su8;

__device__ __forceinline__ v8f zero8() {
    v8f z;
#pragma unroll
    for (int i = 0; i < 8; ++i) z[i] = 0.f;
    return z;
}

__device__ __forceinline__ v8f wmma_f16(v16h a, v16h b, v8f c) {
    // (neg_a, A, neg_b, B, c_mod, C, reuse_a, reuse_b)
    return __builtin_amdgcn_wmma_f32_16x16x32_f16(false, a, false, b, (short)0, c,
                                                  false, false);
}

// A fragment (16x32 f16, MxK) from LDS tile stored row-major [m][k], row
// stride `ld` halves (multiple of 8 => 16B aligned). ISA layout: lane group
// kg=lane>>4 holds K = kg*8+0..7 (halves 0..7) and 16+kg*8+0..7 (halves 8..15).
__device__ __forceinline__ v16h load_frag_a(const _Float16* p, int ld) {
    const int lane = threadIdx.x & 31;
    const int m  = lane & 15;
    const int kg = lane >> 4;
    const _Float16* r = p + m * ld + kg * 8;
    v8h lo = *(const v8h*)(r);
    v8h hi = *(const v8h*)(r + 16);
    v16h o;
#pragma unroll
    for (int i = 0; i < 8; ++i) { o[i] = lo[i]; o[i + 8] = hi[i]; }
    return o;
}

// B fragment (32x16 f16, KxN) from LDS holding B^T row-major: bt[n][k],
// row stride `ld` halves. Lane group kg holds K = kg*8+0..7 (halves 0..7) and
// 16+kg*8+0..7 (halves 8..15) for N=lane&15, matching load_frag_a.
__device__ __forceinline__ v16h load_frag_bt(const _Float16* p, int ld) {
    const int lane = threadIdx.x & 31;
    const int n  = lane & 15;
    const int kg = lane >> 4;
    const _Float16* r = p + n * ld + kg * 8;
    v8h lo = *(const v8h*)(r);
    v8h hi = *(const v8h*)(r + 16);
    v16h o;
#pragma unroll
    for (int i = 0; i < 8; ++i) { o[i] = lo[i]; o[i + 8] = hi[i]; }
    return o;
}

#if USE_TDM
// Issue a TDM 2D tile load (global f16 -> LDS) per CDNA5 ISA.
// D# group0: [1:0]=count=1, [63:32]=lds_addr, [120:64]=global_addr, [127:126]=type(2).
// D# group1: [17:16]=data_size(1=2B), [20]=pad_enable, [24:22]=pad_interval,
//            [31:25]=pad_amount, [79:48]=tensor_dim0, [111:80]=tensor_dim1,
//            [127:112]=tile_dim0, [143:128]=tile_dim1, [207:160]=dim0_stride.
// pad codes: interval k -> (2<<k) DWORDs between pads; amount k -> (k+1) DWORDs.
// Call with a wave-uniform guard (EXEC is ignored by TDM; one issue per wave).
__device__ __forceinline__ void tdm_load_2d_f16(unsigned int lds_byte_addr,
                                                const _Float16* gsrc,
                                                unsigned int tile0,
                                                unsigned int tile1,
                                                unsigned int stride_elems,
                                                unsigned int pad_int_code,
                                                unsigned int pad_amt_code) {
    const unsigned long long ga = (unsigned long long)(uintptr_t)gsrc;
    su4 g0;
    g0[0] = 1u;                                                // count=1, user D#
    g0[1] = lds_byte_addr;
    g0[2] = (unsigned int)ga;
    g0[3] = ((unsigned int)(ga >> 32) & 0x01FFFFFFu) | (2u << 30);  // type=2
    su8 g1;
    g1[0] = (1u << 16) | (1u << 20) | (pad_int_code << 22) | (pad_amt_code << 25);
    g1[1] = (tile0 & 0xFFFFu) << 16;                           // tensor_dim0 lo16
    g1[2] = (tile0 >> 16) | ((tile1 & 0xFFFFu) << 16);         // dim0 hi / dim1 lo
    g1[3] = (tile1 >> 16) | (tile0 << 16);                     // dim1 hi / tile_dim0
    g1[4] = tile1 & 0xFFFFu;                                   // tile_dim1, tile_dim2=0
    g1[5] = stride_elems;                                      // dim0_stride lo32
    g1[6] = 0u;                                                // dim0_stride hi16
    g1[7] = 0u;
    asm volatile("tensor_load_to_lds %0, %1" :: "s"(g0), "s"(g1) : "memory");
}
#endif

// ---------------------------------------------------------------------------
// RMSNorm over channel dim of x [B,C,H,W] -> xn f16 [NPIX][C] (pixel-major)
// ---------------------------------------------------------------------------
__global__ void __launch_bounds__(128) rms_x_kernel(const float* __restrict__ x,
                                                    const float* __restrict__ g,
                                                    _Float16* __restrict__ xn) {
    const int pix = blockIdx.x;            // 0..NPIX-1
    const int b = pix >> 10, hw = pix & 1023;
    const int t = threadIdx.x;
    float vals[4];
    float ss = 0.f;
#pragma unroll
    for (int i = 0; i < 4; ++i) {
        const int c = t + i * 128;
        vals[i] = x[((size_t)(b * C_ + c)) * HW_ + hw];
        ss += vals[i] * vals[i];
    }
    __shared__ float red[128];
    red[t] = ss;
    __syncthreads();
    for (int s = 64; s > 0; s >>= 1) {
        if (t < s) red[t] += red[t + s];
        __syncthreads();
    }
    const float rs = rsqrtf(red[0] * (1.f / C_) + EPS_);
#pragma unroll
    for (int i = 0; i < 4; ++i) {
        const int c = t + i * 128;
        xn[(size_t)pix * C_ + c] = (_Float16)(vals[i] * rs * g[c]);
    }
}

// RMSNorm over last dim of context [B,Lc,C] -> cn f16 [NTOK][C]
__global__ void __launch_bounds__(128) rms_ctx_kernel(const float* __restrict__ ctx,
                                                      const float* __restrict__ g,
                                                      _Float16* __restrict__ cn) {
    const int tok = blockIdx.x;
    const int t = threadIdx.x;
    float vals[4];
    float ss = 0.f;
#pragma unroll
    for (int i = 0; i < 4; ++i) {
        const int c = t + i * 128;
        vals[i] = ctx[(size_t)tok * C_ + c];
        ss += vals[i] * vals[i];
    }
    __shared__ float red[128];
    red[t] = ss;
    __syncthreads();
    for (int s = 64; s > 0; s >>= 1) {
        if (t < s) red[t] += red[t + s];
        __syncthreads();
    }
    const float rs = rsqrtf(red[0] * (1.f / C_) + EPS_);
#pragma unroll
    for (int i = 0; i < 4; ++i) {
        const int c = t + i * 128;
        cn[(size_t)tok * C_ + c] = (_Float16)(vals[i] * rs * g[c]);
    }
}

// W [O][C] f32 -> Wt [C][O] f16 (B^T layout for TN GEMM)
__global__ void __launch_bounds__(256) transpose_w_kernel(const float* __restrict__ W,
                                                          _Float16* __restrict__ Wt,
                                                          int O, int C) {
    const int idx = blockIdx.x * 256 + threadIdx.x;
    if (idx < O * C) {
        const int o = idx / C, c = idx % C;
        Wt[(size_t)c * O + o] = (_Float16)W[idx];
    }
}

// ---------------------------------------------------------------------------
// WMMA GEMM: C[M][N] = alpha * A[M][K](f16) * Bt[K][N](f16)  (+ bias on f32 path)
// Block tile 128x128x32, 256 threads = 8 waves in a 4(M)x2(N) grid,
// each wave owns a 32x64 region = 2x4 WMMA accumulators.
// ---------------------------------------------------------------------------
__global__ void __launch_bounds__(256) gemm_tn_kernel(
        const _Float16* __restrict__ A, const _Float16* __restrict__ Bt,
        const float* __restrict__ bias, float alpha, int M, int N, int K,
        _Float16* __restrict__ Ch, float* __restrict__ Cf) {
    constexpr int LDA = 40;   // 32 + 8 pad halves (80B rows -> 16B aligned)
    constexpr int LDB = 40;
    __shared__ __attribute__((aligned(16))) _Float16 sA[128 * LDA];
    __shared__ __attribute__((aligned(16))) _Float16 sB[128 * LDB]; // B^T: [n][k]
    const int t = threadIdx.x;
    const int lane = t & 31, wid = t >> 5;
    const int wm = wid & 3, wn = wid >> 2;
    const int m0 = blockIdx.y * 128, n0 = blockIdx.x * 128;

    v8f acc[2][4];
#pragma unroll
    for (int tm = 0; tm < 2; ++tm)
#pragma unroll
        for (int tn = 0; tn < 4; ++tn) acc[tm][tn] = zero8();

    for (int k0 = 0; k0 < K; k0 += 32) {
        // Stage A tile (128x32): thread loads 16 contiguous halves.
        {
            const int row = t >> 1, ko = (t & 1) * 16;
            const _Float16* src = A + (size_t)(m0 + row) * K + k0 + ko;
            uint4 d0 = *(const uint4*)(src);
            uint4 d1 = *(const uint4*)(src + 8);
            *(uint4*)(&sA[row * LDA + ko])     = d0;
            *(uint4*)(&sA[row * LDA + ko + 8]) = d1;
        }
        // Stage B tile transposed into LDS: global Bt is [K][N] row-major.
        {
            const int k = t >> 3, no = (t & 7) * 16;
            const _Float16* src = Bt + (size_t)(k0 + k) * N + n0 + no;
            union { uint4 u; _Float16 h[8]; } u0, u1;
            u0.u = *(const uint4*)(src);
            u1.u = *(const uint4*)(src + 8);
#pragma unroll
            for (int i = 0; i < 8; ++i) sB[(no + i) * LDB + k] = u0.h[i];
#pragma unroll
            for (int i = 0; i < 8; ++i) sB[(no + 8 + i) * LDB + k] = u1.h[i];
        }
        __syncthreads();
        if (k0 + 32 < K) {  // hint next tiles into cache (global_prefetch_b8)
            __builtin_prefetch(A + (size_t)(m0 + (t >> 1)) * K + k0 + 32 + (t & 1) * 16, 0, 1);
            __builtin_prefetch(Bt + (size_t)(k0 + 32 + (t >> 3)) * N + n0 + (t & 7) * 16, 0, 1);
        }
        v16h af[2];
#pragma unroll
        for (int tm = 0; tm < 2; ++tm)
            af[tm] = load_frag_a(&sA[(wm * 32 + tm * 16) * LDA], LDA);
#pragma unroll
        for (int tn = 0; tn < 4; ++tn) {
            v16h bf = load_frag_bt(&sB[(wn * 64 + tn * 16) * LDB], LDB);
#pragma unroll
            for (int tm = 0; tm < 2; ++tm)
                acc[tm][tn] = wmma_f16(af[tm], bf, acc[tm][tn]);
        }
        __syncthreads();
    }

    // Epilogue. C fragment layout: lane holds column n=lane&15, rows r+8*(lane>>4).
    const int n = lane & 15, hi = lane >> 4;
#pragma unroll
    for (int tm = 0; tm < 2; ++tm)
#pragma unroll
        for (int tn = 0; tn < 4; ++tn) {
            const int col = n0 + wn * 64 + tn * 16 + n;
#pragma unroll
            for (int r = 0; r < 8; ++r) {
                const int row = m0 + wm * 32 + tm * 16 + r + 8 * hi;
                const float v = acc[tm][tn][r] * alpha;
                if (Cf) Cf[(size_t)row * N + col] = v + (bias ? bias[col] : 0.f);
                else    Ch[(size_t)row * N + col] = (_Float16)v;
            }
        }
}

// ---------------------------------------------------------------------------
// Attention: per (b, head, 64-query tile). Keys processed in chunks of 128.
// Q and K tiles staged by the Tensor Data Mover (hardware LDS row padding
// reproduces the +8-half pad). Softmax without max subtraction (scores are
// O(1): q pre-scaled by 1/8 on RMS-normalized data) -> only row-sums tracked.
//   S = Q(64x64) * K^T  -> P = exp(S) -> O += P * V ; O /= rowsum at the end.
// ---------------------------------------------------------------------------
#define LDQ_ 72     // 64 + 8 pad halves (32 DWORDs data + 4 DWORDs pad)
#define LDK_ 72
#define LDV_ 136    // 128 + 8
#define LDP_ 136

__global__ void __launch_bounds__(256) attn_kernel(const _Float16* __restrict__ qb,
                                                   const _Float16* __restrict__ kvb,
                                                   _Float16* __restrict__ ob) {
    __shared__ __attribute__((aligned(16))) _Float16 sQ[64 * LDQ_];
    __shared__ __attribute__((aligned(16))) _Float16 sK[128 * LDK_];   // [token][d]
    __shared__ __attribute__((aligned(16))) _Float16 sVT[64 * LDV_];   // [d][token]
    __shared__ __attribute__((aligned(16))) _Float16 sP[64 * LDP_];    // [q][token]
    __shared__ float rowsum[64];

    const int blk = blockIdx.x;
    const int qt = blk & 15;
    const int h  = (blk >> 4) & 7;
    const int b  = blk >> 7;
    const int t = threadIdx.x;
    const int lane = t & 31, wid = t >> 5;
    const int mt = wid & 3;        // query sub-tile (16 rows)
    const int grp = wid >> 2;      // S phase: j-half; O phase: d-half

    if (t < 64) rowsum[t] = 0.f;

    // Stage Q tile 64x64 (rows b*1024+qt*64.., cols h*64..).
#if USE_TDM
    if (wid == 0) {
        tdm_load_2d_f16((unsigned int)(uintptr_t)(void*)sQ,
                        qb + ((size_t)(b * HW_ + qt * 64)) * HID_ + h * HD_,
                        /*tile0=*/64, /*tile1=*/64, /*stride=*/HID_,
                        /*pad_interval: 2<<4=32 DW*/ 4, /*pad_amount: 4 DW*/ 3);
    }
#else
    {
        const int row = t >> 2, doff = (t & 3) * 16;
        const _Float16* src = qb + ((size_t)(b * HW_ + qt * 64 + row)) * HID_ + h * HD_ + doff;
        uint4 d0 = *(const uint4*)(src);
        uint4 d1 = *(const uint4*)(src + 8);
        *(uint4*)(&sQ[row * LDQ_ + doff])     = d0;
        *(uint4*)(&sQ[row * LDQ_ + doff + 8]) = d1;
    }
#endif

    v8f oacc[2] = { zero8(), zero8() };   // this wave's two 16-wide d tiles

    for (int j0 = 0; j0 < LC_; j0 += 128) {
        // Stage K chunk [128 tokens][64 d] (already the B^T layout for Q*K^T).
#if USE_TDM
        if (wid == 0) {
            tdm_load_2d_f16((unsigned int)(uintptr_t)(void*)sK,
                            kvb + ((size_t)(b * LC_ + j0)) * (2 * HID_) + h * HD_,
                            /*tile0=*/64, /*tile1=*/128, /*stride=*/2 * HID_, 4, 3);
        }
#else
        {
            const int tok = t >> 1, doff = (t & 1) * 32;
            const _Float16* src = kvb + ((size_t)(b * LC_ + j0 + tok)) * (2 * HID_)
                                + h * HD_ + doff;
#pragma unroll
            for (int i = 0; i < 32; i += 8)
                *(uint4*)(&sK[tok * LDK_ + doff + i]) = *(const uint4*)(src + i);
        }
#endif
        // Stage V chunk transposed: sVT[d][token] (B^T layout for P*V).
        {
            const int tok = t >> 1, doff = (t & 1) * 32;
            const _Float16* src = kvb + ((size_t)(b * LC_ + j0 + tok)) * (2 * HID_)
                                + HID_ + h * HD_ + doff;
#pragma unroll
            for (int i = 0; i < 32; i += 8) {
                union { uint4 u; _Float16 hh[8]; } u;
                u.u = *(const uint4*)(src + i);
#pragma unroll
                for (int e = 0; e < 8; ++e)
                    sVT[(doff + i + e) * LDV_ + tok] = u.hh[e];
            }
        }
#if USE_TDM
        if (wid == 0) __builtin_amdgcn_s_wait_tensorcnt(0);  // Q (+K) complete
#endif
        __syncthreads();

        // ---- S phase: wave (mt, grp) computes S for 16 queries x 64 keys ----
        v16h aq[2];
#pragma unroll
        for (int ks = 0; ks < 2; ++ks)
            aq[ks] = load_frag_a(&sQ[(mt * 16) * LDQ_ + ks * 32], LDQ_);

        float psum[8];
#pragma unroll
        for (int r = 0; r < 8; ++r) psum[r] = 0.f;

        const int n = lane & 15, hi = lane >> 4;
#pragma unroll
        for (int tj = 0; tj < 4; ++tj) {
            const int jb = grp * 64 + tj * 16;
            v8f s = zero8();
#pragma unroll
            for (int ks = 0; ks < 2; ++ks) {
                v16h bk = load_frag_bt(&sK[jb * LDK_ + ks * 32], LDK_);
                s = wmma_f16(aq[ks], bk, s);
            }
#pragma unroll
            for (int r = 0; r < 8; ++r) {
                const float e = __expf(s[r]);
                psum[r] += e;
                sP[(mt * 16 + r + 8 * hi) * LDP_ + jb + n] = (_Float16)e;
            }
        }
        // Row-sum: reduce over the 16 lanes holding one column each, then
        // accumulate across waves/chunks via LDS float atomics (ds_add_f32).
#pragma unroll
        for (int r = 0; r < 8; ++r) {
            float v = psum[r];
            v += __shfl_xor(v, 1, 16);
            v += __shfl_xor(v, 2, 16);
            v += __shfl_xor(v, 4, 16);
            v += __shfl_xor(v, 8, 16);
            if ((lane & 15) == 0)
                atomicAdd(&rowsum[mt * 16 + r + 8 * hi], v);
        }
        __syncthreads();   // P complete before O accumulation

        // ---- O phase: wave (mt, grp) accumulates 16 queries x 32 d-cols ----
#pragma unroll
        for (int ks = 0; ks < 4; ++ks) {
            v16h ap = load_frag_a(&sP[(mt * 16) * LDP_ + ks * 32], LDP_);
#pragma unroll
            for (int dt = 0; dt < 2; ++dt) {
                const int dbase = (grp * 2 + dt) * 16;
                v16h bv = load_frag_bt(&sVT[dbase * LDV_ + ks * 32], LDV_);
                oacc[dt] = wmma_f16(ap, bv, oacc[dt]);
            }
        }
        __syncthreads();   // before next chunk overwrites sK/sVT/sP
    }

    // Normalize by rowsum and store as f16 [pixel][head*64+d].
    const int n = lane & 15, hi = lane >> 4;
#pragma unroll
    for (int dt = 0; dt < 2; ++dt) {
        const int dbase = (grp * 2 + dt) * 16;
#pragma unroll
        for (int r = 0; r < 8; ++r) {
            const int row = mt * 16 + r + 8 * hi;
            const float inv = 1.f / rowsum[row];
            ob[((size_t)(b * HW_ + qt * 64 + row)) * HID_ + h * HD_ + dbase + n] =
                (_Float16)(oacc[dt][r] * inv);
        }
    }
}

// ---------------------------------------------------------------------------
// Fused: channel RMSNorm of y (bias already added in GEMM) + residual.
// yraw: [NPIX][C] f32 (pixel-major); out: [B,C,H,W] f32.
// ---------------------------------------------------------------------------
__global__ void __launch_bounds__(128) final_kernel(const float* __restrict__ yraw,
                                                    const float* __restrict__ x,
                                                    const float* __restrict__ g2,
                                                    float* __restrict__ out) {
    const int pix = blockIdx.x;
    const int b = pix >> 10, hw = pix & 1023;
    const int t = threadIdx.x;
    float vals[4];
    float ss = 0.f;
#pragma unroll
    for (int i = 0; i < 4; ++i) {
        const int c = t + i * 128;
        vals[i] = yraw[(size_t)pix * C_ + c];
        ss += vals[i] * vals[i];
    }
    __shared__ float red[128];
    red[t] = ss;
    __syncthreads();
    for (int s = 64; s > 0; s >>= 1) {
        if (t < s) red[t] += red[t + s];
        __syncthreads();
    }
    const float rs = rsqrtf(red[0] * (1.f / C_) + EPS_);
#pragma unroll
    for (int i = 0; i < 4; ++i) {
        const int c = t + i * 128;
        const size_t xi = ((size_t)(b * C_ + c)) * HW_ + hw;
        out[xi] = vals[i] * rs * g2[c] + x[xi];
    }
}

// ---------------------------------------------------------------------------
extern "C" void kernel_launch(void* const* d_in, const int* in_sizes, int n_in,
                              void* d_out, int out_size, void* d_ws, size_t ws_size,
                              hipStream_t stream) {
    (void)in_sizes; (void)n_in; (void)out_size; (void)ws_size;
    const float* x   = (const float*)d_in[0];
    const float* ctx = (const float*)d_in[1];
    const float* Wq  = (const float*)d_in[2];
    const float* Wkv = (const float*)d_in[3];
    const float* Wo  = (const float*)d_in[4];
    const float* bo  = (const float*)d_in[5];
    const float* g   = (const float*)d_in[6];
    const float* g2  = (const float*)d_in[7];
    float* out = (float*)d_out;

    char* ws = (char*)d_ws;
    size_t off = 0;
    auto carve = [&](size_t bytes) {
        char* p = ws + off;
        off += (bytes + 255) & ~(size_t)255;
        return p;
    };
    _Float16* xn   = (_Float16*)carve((size_t)NPIX_ * C_ * 2);
    _Float16* cn   = (_Float16*)carve((size_t)NTOK_ * C_ * 2);
    _Float16* WqT  = (_Float16*)carve((size_t)C_ * HID_ * 2);
    _Float16* WkvT = (_Float16*)carve((size_t)C_ * 2 * HID_ * 2);
    _Float16* WoT  = (_Float16*)carve((size_t)HID_ * C_ * 2);
    _Float16* qb   = (_Float16*)carve((size_t)NPIX_ * HID_ * 2);
    _Float16* kvb  = (_Float16*)carve((size_t)NTOK_ * 2 * HID_ * 2);
    _Float16* ao   = (_Float16*)carve((size_t)NPIX_ * HID_ * 2);
    float*    yraw = (float*)carve((size_t)NPIX_ * C_ * 4);

    rms_x_kernel<<<NPIX_, 128, 0, stream>>>(x, g, xn);
    rms_ctx_kernel<<<NTOK_, 128, 0, stream>>>(ctx, g, cn);
    transpose_w_kernel<<<(HID_ * C_) / 256, 256, 0, stream>>>(Wq, WqT, HID_, C_);
    transpose_w_kernel<<<(2 * HID_ * C_) / 256, 256, 0, stream>>>(Wkv, WkvT, 2 * HID_, C_);
    transpose_w_kernel<<<(C_ * HID_) / 256, 256, 0, stream>>>(Wo, WoT, C_, HID_);

    // Q = xn * WqT, scaled by HEAD_DIM^-0.5 = 0.125
    gemm_tn_kernel<<<dim3(HID_ / 128, NPIX_ / 128), 256, 0, stream>>>(
        xn, WqT, nullptr, 0.125f, NPIX_, HID_, C_, qb, nullptr);
    // KV = cn * WkvT
    gemm_tn_kernel<<<dim3(2 * HID_ / 128, NTOK_ / 128), 256, 0, stream>>>(
        cn, WkvT, nullptr, 1.0f, NTOK_, 2 * HID_, C_, kvb, nullptr);
    // Attention: B*HEADS*(HW/64) = 1024 blocks
    attn_kernel<<<B_ * HEADS_ * (HW_ / 64), 256, 0, stream>>>(qb, kvb, ao);
    // y = ao * WoT + bo (f32 out)
    gemm_tn_kernel<<<dim3(C_ / 128, NPIX_ / 128), 256, 0, stream>>>(
        ao, WoT, bo, 1.0f, NPIX_, C_, HID_, nullptr, yraw);
    // rmsnorm + residual
    final_kernel<<<NPIX_, 128, 0, stream>>>(yraw, x, g2, out);
}
